// Model_12704513261975
// MI455X (gfx1250) — matrix operand order repaired and checked by reading, not compiled, not run on hardware
//
#include <hip/hip_runtime.h>
#include <hip/hip_bf16.h>

// ---------------------------------------------------------------------------
// CDNA5 WMMA types (wave32, v_wmma_f32_16x16x32_bf16)
// ---------------------------------------------------------------------------
typedef __attribute__((ext_vector_type(16))) __bf16 v16bf;
typedef __attribute__((ext_vector_type(8)))  float  v8f;
typedef __attribute__((ext_vector_type(4)))  int    v4i;

union FragAB { v16bf v; unsigned u[8]; };
union FragC  { v8f   v; float    f[8]; };

#if __has_builtin(__builtin_amdgcn_global_load_async_to_lds_b128) && \
    __has_builtin(__builtin_amdgcn_s_wait_asynccnt)
#define HAS_ASYNC 1
#else
#define HAS_ASYNC 0
#endif

__device__ __forceinline__ unsigned short f2bf(float f) {
    unsigned u = __builtin_bit_cast(unsigned, f);
    return (unsigned short)((u + 0x7FFFu + ((u >> 16) & 1u)) >> 16); // RNE
}
__device__ __forceinline__ unsigned pack_bf16x2(float lo, float hi) {
#if __has_builtin(__builtin_amdgcn_cvt_pk_bf16_f32)
    auto pk = __builtin_amdgcn_cvt_pk_bf16_f32(lo, hi);   // v_cvt_pk_bf16_f32
    unsigned r; __builtin_memcpy(&r, &pk, 4);
    return r;
#else
    return (unsigned)f2bf(lo) | ((unsigned)f2bf(hi) << 16);
#endif
}
__device__ __forceinline__ float silu_f(float x)     { return x / (1.f + expf(-x)); }
__device__ __forceinline__ float softplus_f(float x) { return (x > 20.f) ? x : log1pf(expf(x)); }

#define LDP 20  // padded row stride in dwords for packed bf16 tiles

// ---- shared helper: build fragments from packed tiles, 4 wmma --------------
__device__ __forceinline__ void frag_mma(const unsigned (*As)[LDP],
                                         const unsigned (*Bs)[LDP],
                                         int wm, int wn, int l16, int hi,
                                         FragC acc[4])
{
    FragAB a, b0, b1, b2, b3;
    const int ar = wm * 16 + l16;
    const int bn = wn * 64 + l16;
    #pragma unroll
    for (int p = 0; p < 8; ++p) {
        // A 16x32 bf16: pair index = (p<4 ? p : p+4) + 4*hi
        int pa = ((p < 4) ? p : p + 4) + 4 * hi;
        a.u[p] = As[ar][pa];
        // B 32x16 bf16: pair index = (p<4 ? p : p+4) + 4*hi
        int pb = ((p < 4) ? p : p + 4) + 4 * hi;
        b0.u[p] = Bs[bn][pb];
        b1.u[p] = Bs[bn + 16][pb];
        b2.u[p] = Bs[bn + 32][pb];
        b3.u[p] = Bs[bn + 48][pb];
    }
    acc[0].v = __builtin_amdgcn_wmma_f32_16x16x32_bf16(false, a.v, false, b0.v,
                                                       (short)0, acc[0].v, false, false);
    acc[1].v = __builtin_amdgcn_wmma_f32_16x16x32_bf16(false, a.v, false, b1.v,
                                                       (short)0, acc[1].v, false, false);
    acc[2].v = __builtin_amdgcn_wmma_f32_16x16x32_bf16(false, a.v, false, b2.v,
                                                       (short)0, acc[2].v, false, false);
    acc[3].v = __builtin_amdgcn_wmma_f32_16x16x32_bf16(false, a.v, false, b3.v,
                                                       (short)0, acc[3].v, false, false);
}

#if HAS_ASYNC
// ---- async helpers ---------------------------------------------------------
__device__ __forceinline__ void issue_tile(const float* __restrict__ A, int lda,
                                           const float* __restrict__ W, int K,
                                           int m0, int n0, int kk, int t,
                                           float (*Af)[32], float (*Bf)[32])
{
    #pragma unroll
    for (int i = 0; i < 2; ++i) {               // A: 512 float4
        int f = t + i * 256;
        int r = f >> 3, c4 = (f & 7) << 2;
        const float* gp = &A[(size_t)(m0 + r) * lda + kk + c4];
        __builtin_amdgcn_global_load_async_to_lds_b128(
            (__attribute__((address_space(1))) v4i*)(void*)gp,
            (__attribute__((address_space(3))) v4i*)(void*)&Af[r][c4], 0, 0);
    }
    #pragma unroll
    for (int i = 0; i < 4; ++i) {               // B: 1024 float4
        int f = t + i * 256;
        int r = f >> 3, c4 = (f & 7) << 2;
        const float* gp = &W[(size_t)(n0 + r) * K + kk + c4];
        __builtin_amdgcn_global_load_async_to_lds_b128(
            (__attribute__((address_space(1))) v4i*)(void*)gp,
            (__attribute__((address_space(3))) v4i*)(void*)&Bf[r][c4], 0, 0);
    }
}
__device__ __forceinline__ void convert_tile(int t,
                                             const float (*Af)[32], const float (*Bf)[32],
                                             unsigned (*As)[LDP], unsigned (*Bs)[LDP])
{
    #pragma unroll
    for (int i = 0; i < 4; ++i) {               // A: 1024 pairs
        int f = t + i * 256;
        int r = f >> 4, pc = f & 15;
        As[r][pc] = pack_bf16x2(Af[r][2 * pc], Af[r][2 * pc + 1]);
    }
    #pragma unroll
    for (int i = 0; i < 8; ++i) {               // B: 2048 pairs
        int f = t + i * 256;
        int r = f >> 4, pc = f & 15;
        Bs[r][pc] = pack_bf16x2(Bf[r][2 * pc], Bf[r][2 * pc + 1]);
    }
}
#endif

// ---------------------------------------------------------------------------
// GEMM: C(M,N) = A(M,K) @ W(N,K)^T + bias(N) [+ add(M,N)]
// Block: 256 threads (8 waves), tile 64(M) x 128(N), K staged 32 at a time.
// Wave (wm = w&3, wn = w>>2) computes 16x64: 1 A-frag x 4 B-frags = 4 wmma/k-step.
// ASYNC=true (host guarantees K%32==0, N%128==0, lda%4==0): double-buffered
// fp32 staging via global_load_async_to_lds_b128; tile k+1 is issued before
// waiting on tile k (s_wait_asynccnt 6 — async loads complete in order).
// Requires M % 64 == 0 (true here: M = 4096).
// ---------------------------------------------------------------------------
template<bool ASYNC>
__global__ __launch_bounds__(256) void gemm_wmma_kernel(
    const float* __restrict__ A, int lda,
    const float* __restrict__ W,               // (N,K) row-major, ld = K
    const float* __restrict__ bias,            // nullable
    const float* __restrict__ addp, int ldadd, // nullable
    float* __restrict__ C, int ldc,
    int M, int N, int K)
{
    __shared__ unsigned As[64][LDP];                   // packed bf16 pairs
    __shared__ unsigned Bs[128][LDP];
    __shared__ float Af[ASYNC ? 2 : 1][64][32];        // fp32 staging (async)
    __shared__ float Bf[ASYNC ? 2 : 1][128][32];

    const int t    = threadIdx.x;
    const int lane = t & 31;
    const int wave = t >> 5;
    const int wm   = wave & 3;
    const int wn   = wave >> 2;
    const int m0   = blockIdx.y * 64;
    const int n0   = blockIdx.x * 128;
    const int hi   = lane >> 4;
    const int l16  = lane & 15;

    FragC acc[4];
    #pragma unroll
    for (int j = 0; j < 4; ++j)
        #pragma unroll
        for (int i = 0; i < 8; ++i) acc[j].f[i] = 0.f;

    if constexpr (ASYNC) {
#if HAS_ASYNC
        int buf = 0;
        issue_tile(A, lda, W, K, m0, n0, 0, t, Af[0], Bf[0]);
        for (int k0 = 0; k0 < K; k0 += 32) {
            const int nxt = k0 + 32;
            if (nxt < K) {
                issue_tile(A, lda, W, K, m0, n0, nxt, t, Af[buf ^ 1], Bf[buf ^ 1]);
                __builtin_amdgcn_s_wait_asynccnt(6);   // tile k's 6 loads done
            } else {
                __builtin_amdgcn_s_wait_asynccnt(0);
            }
            __syncthreads();
            convert_tile(t, Af[buf], Bf[buf], As, Bs);
            __syncthreads();
            frag_mma(As, Bs, wm, wn, l16, hi, acc);
            __syncthreads();
            buf ^= 1;
        }
#endif
    } else {
        for (int k0 = 0; k0 < K; k0 += 32) {
            // ---- sync stage: branchless (clamped addresses + value selects)
            const bool kfull = (k0 + 32 <= K);
            if (kfull && ((lda & 3) == 0)) {
                #pragma unroll
                for (int i = 0; i < 2; ++i) {
                    int f = t + i * 256;
                    int r = f >> 3, c4 = (f & 7) << 2;
                    float4 v = *reinterpret_cast<const float4*>(
                        &A[(size_t)(m0 + r) * lda + k0 + c4]);
                    As[r][c4 >> 1]       = pack_bf16x2(v.x, v.y);
                    As[r][(c4 >> 1) + 1] = pack_bf16x2(v.z, v.w);
                }
            } else {
                #pragma unroll
                for (int i = 0; i < 4; ++i) {
                    int f = t + i * 256;
                    int r = f >> 4, pc = f & 15;
                    int k = k0 + 2 * pc;
                    const float* row = &A[(size_t)(m0 + r) * lda];
                    float a0 = row[(k     < K) ? k     : (K - 1)];
                    float a1 = row[(k + 1 < K) ? k + 1 : (K - 1)];
                    a0 = (k     < K) ? a0 : 0.f;
                    a1 = (k + 1 < K) ? a1 : 0.f;
                    As[r][pc] = pack_bf16x2(a0, a1);
                }
            }
            if (kfull && ((K & 3) == 0)) {
                #pragma unroll
                for (int i = 0; i < 4; ++i) {
                    int f = t + i * 256;
                    int r = f >> 3, c4 = (f & 7) << 2;
                    int nn = n0 + r;
                    int nc = (nn < N) ? nn : (N - 1);
                    float4 v = *reinterpret_cast<const float4*>(
                        &W[(size_t)nc * K + k0 + c4]);
                    if (nn >= N) { v.x = 0.f; v.y = 0.f; v.z = 0.f; v.w = 0.f; }
                    Bs[r][c4 >> 1]       = pack_bf16x2(v.x, v.y);
                    Bs[r][(c4 >> 1) + 1] = pack_bf16x2(v.z, v.w);
                }
            } else {
                #pragma unroll
                for (int i = 0; i < 8; ++i) {
                    int f = t + i * 256;
                    int r = f >> 4, pc = f & 15;
                    int nn = n0 + r;
                    int nc = (nn < N) ? nn : (N - 1);
                    int k = k0 + 2 * pc;
                    const float* row = &W[(size_t)nc * K];
                    float b0 = row[(k     < K) ? k     : (K - 1)];
                    float b1 = row[(k + 1 < K) ? k + 1 : (K - 1)];
                    b0 = (k     < K && nn < N) ? b0 : 0.f;
                    b1 = (k + 1 < K && nn < N) ? b1 : 0.f;
                    Bs[r][pc] = pack_bf16x2(b0, b1);
                }
            }
            __syncthreads();
            frag_mma(As, Bs, wm, wn, l16, hi, acc);
            __syncthreads();
        }
    }

    // ---- epilogue: lane -> n = lane%16 ; VGPR r -> m = r + 8*(lane/16)
    // Uniform conditions hoisted out of the store loops.
    const int mb = m0 + wm * 16 + hi * 8;
    #pragma unroll
    for (int j = 0; j < 4; ++j) {
        int nc = n0 + wn * 64 + j * 16 + l16;
        if (nc >= N) continue;
        float bv = bias ? bias[nc] : 0.f;
        if (addp) {
            #pragma unroll
            for (int r = 0; r < 8; ++r) {
                int m = mb + r;
                C[(size_t)m * ldc + nc] = acc[j].f[r] + bv + addp[(size_t)m * ldadd + nc];
            }
        } else {
            #pragma unroll
            for (int r = 0; r < 8; ++r) {
                int m = mb + r;
                C[(size_t)m * ldc + nc] = acc[j].f[r] + bv;
            }
        }
    }
}

// ---------------------------------------------------------------------------
// RevIN-style normalization: x (B,L,C) -> xn (B*C, L), per-(b,c) mean/std.
// ---------------------------------------------------------------------------
__global__ __launch_bounds__(256) void normalize_kernel(
    const float* __restrict__ x, float* __restrict__ xn,
    float* __restrict__ meanv, float* __restrict__ stdv,
    int Bb, int L, int C)
{
    int bc = blockIdx.x;
    int b = bc / C, c = bc % C;
    __shared__ float s1[256], s2[256];
    float sum = 0.f, sq = 0.f;
    for (int l = threadIdx.x; l < L; l += 256) {
        float v = x[((size_t)b * L + l) * C + c];
        sum += v; sq += v * v;
    }
    s1[threadIdx.x] = sum; s2[threadIdx.x] = sq;
    __syncthreads();
    for (int s = 128; s > 0; s >>= 1) {
        if (threadIdx.x < s) { s1[threadIdx.x] += s1[threadIdx.x + s]; s2[threadIdx.x] += s2[threadIdx.x + s]; }
        __syncthreads();
    }
    float mu  = s1[0] / (float)L;
    float var = s2[0] / (float)L - mu * mu;
    float sd  = sqrtf(var + 1e-5f);
    if (threadIdx.x == 0) { meanv[bc] = mu; stdv[bc] = sd; }
    float inv = 1.f / sd;
    for (int l = threadIdx.x; l < L; l += 256)
        xn[(size_t)bc * L + l] = (x[((size_t)b * L + l) * C + c] - mu) * inv;
}

// ---------------------------------------------------------------------------
// Mamba (seq-len 1) elementwise pieces.
// ---------------------------------------------------------------------------
__global__ void mamba_pre_kernel(const float* __restrict__ xz,
                                 const float* __restrict__ conv_w,
                                 const float* __restrict__ conv_b,
                                 float* __restrict__ xc, int R, int di)
{
    int i = blockIdx.x * blockDim.x + threadIdx.x;
    if (i >= R * di) return;
    int r = i / di, d = i - r * di;
    float v = conv_w[2 * d + 1] * xz[(size_t)r * 2 * di + d] + conv_b[d];
    xc[i] = silu_f(v);
}

__global__ void bcdot_kernel(const float* __restrict__ dbl, int ld,
                             int offB, int offC, float* __restrict__ bc, int R)
{
    int r = blockIdx.x * blockDim.x + threadIdx.x;
    if (r >= R) return;
    const float* row = dbl + (size_t)r * ld;
    float s = 0.f;
    #pragma unroll
    for (int n = 0; n < 16; ++n) s += row[offB + n] * row[offC + n];
    bc[r] = s;
}

__global__ void mamba_post_kernel(const float* __restrict__ dtraw,
                                  const float* __restrict__ xc,
                                  const float* __restrict__ xz,
                                  const float* __restrict__ bc,
                                  const float* __restrict__ Dp,
                                  float* __restrict__ u, int R, int di)
{
    int i = blockIdx.x * blockDim.x + threadIdx.x;
    if (i >= R * di) return;
    int r = i / di, d = i - r * di;
    float dt  = softplus_f(dtraw[i]);
    float xcv = xc[i];
    float z   = xz[(size_t)r * 2 * di + di + d];
    u[i] = (dt * xcv * bc[r] + Dp[d] * xcv) * silu_f(z);
}

// ---------------------------------------------------------------------------
// Mamba with d_model==1 over a real sequence: one thread per row.
// ---------------------------------------------------------------------------
__global__ void mamba_seq_kernel(const float* __restrict__ seq, int ldin,
                                 float* __restrict__ outp, int ldout, int addmode,
                                 const float* __restrict__ in_w,
                                 const float* __restrict__ conv_w,
                                 const float* __restrict__ conv_b,
                                 const float* __restrict__ x_w,
                                 const float* __restrict__ dt_w,
                                 const float* __restrict__ dt_b,
                                 const float* __restrict__ A_log,
                                 const float* __restrict__ Dp,
                                 const float* __restrict__ out_w,
                                 int R, int L)
{
    int r = blockIdx.x * blockDim.x + threadIdx.x;
    if (r >= R) return;
    float iw0 = in_w[0], iw1 = in_w[1];
    float c0 = conv_w[0], c1 = conv_w[1], cb = conv_b[0];
    float xwdt = x_w[0];
    float xwB[16], xwC[16], An[16], h[16];
    #pragma unroll
    for (int n = 0; n < 16; ++n) {
        xwB[n] = x_w[1 + n];
        xwC[n] = x_w[17 + n];
        An[n]  = -expf(A_log[n]);
        h[n]   = 0.f;
    }
    float dtw = dt_w[0], dtb = dt_b[0], Dv = Dp[0], ow = out_w[0];
    const float* in = seq + (size_t)r * ldin;
    float* out = outp + (size_t)r * ldout;
    float prev = 0.f;
    for (int t = 0; t < L; ++t) {
        float x   = in[t];
        float xcr = iw0 * x;
        float xc  = silu_f(c0 * prev + c1 * xcr + cb);
        prev = xcr;
        float dt = softplus_f(dtw * (xwdt * xc) + dtb);
        float y  = Dv * xc;
        #pragma unroll
        for (int n = 0; n < 16; ++n) {
            float dA = expf(dt * An[n]);
            h[n] = dA * h[n] + dt * (xwB[n] * xc) * xc;
            y += h[n] * (xwC[n] * xc);
        }
        float z = iw1 * x;
        float o = y * silu_f(z) * ow;
        out[t] = addmode ? (out[t] + o) : o;
    }
}

__global__ void add3_kernel(const float* a, const float* b, const float* c,
                            float* o, int n)
{
    int i = blockIdx.x * blockDim.x + threadIdx.x;
    if (i < n) o[i] = a[i] + b[i] + c[i];
}

__global__ void final_kernel(const float* __restrict__ p4,
                             const float* __restrict__ meanv,
                             const float* __restrict__ stdv,
                             float* __restrict__ out, int Bb, int P, int C)
{
    int i = blockIdx.x * blockDim.x + threadIdx.x;
    if (i >= Bb * P * C) return;
    int c = i % C;
    int t = (i / C) % P;
    int b = i / (P * C);
    int r = b * C + c;
    out[i] = p4[(size_t)r * P + t] * stdv[r] + meanv[r];
}

// ---------------------------------------------------------------------------
// Host orchestration
// ---------------------------------------------------------------------------
extern "C" void kernel_launch(void* const* d_in, const int* in_sizes, int n_in,
                              void* d_out, int out_size, void* d_ws, size_t ws_size,
                              hipStream_t stream)
{
    (void)in_sizes; (void)n_in; (void)out_size; (void)ws_size;
    const int Bb = 64, L = 512, C = 64, P = 96;
    const int R = Bb * C; // 4096

    const float* X = (const float*)d_in[0];
    const float *LW[9], *LB[9];
    int p = 1;
    for (int i = 1; i <= 8; ++i) { LW[i] = (const float*)d_in[p++]; LB[i] = (const float*)d_in[p++]; }
    struct MP { const float *in_w, *conv_w, *conv_b, *x_w, *dt_w, *dt_b, *A_log, *D, *out_w; };
    MP M_[9];
    for (int i = 1; i <= 8; ++i) {
        M_[i].in_w   = (const float*)d_in[p++];
        M_[i].conv_w = (const float*)d_in[p++];
        M_[i].conv_b = (const float*)d_in[p++];
        M_[i].x_w    = (const float*)d_in[p++];
        M_[i].dt_w   = (const float*)d_in[p++];
        M_[i].dt_b   = (const float*)d_in[p++];
        M_[i].A_log  = (const float*)d_in[p++];
        M_[i].D      = (const float*)d_in[p++];
        M_[i].out_w  = (const float*)d_in[p++];
    }

    float* w = (float*)d_ws;
    size_t off = 0;
    auto alc = [&](size_t n) { float* q = w + off; off += n; return q; };
    float* xn     = alc((size_t)R * 512);
    float* meanB  = alc(R);
    float* stdB   = alc(R);
    float* h1     = alc((size_t)R * 512);
    float* h2     = alc((size_t)R * 256);
    float* x5     = alc((size_t)R * 128);
    float* x7     = alc((size_t)R * 16);
    float* cat    = alc((size_t)R * 2048);
    float* p4     = alc((size_t)R * 96);
    float* tmpA   = alc((size_t)R * 256);
    float* tmpB   = alc((size_t)R * 256);
    float* hsum   = alc((size_t)R * 256);
    float* s6     = alc((size_t)R * 128);
    float* s8     = alc((size_t)R * 16);
    float* xzb    = alc((size_t)R * 1024);
    float* xcb    = alc((size_t)R * 512);
    float* dblb   = alc((size_t)R * 64);
    float* dtrawb = alc((size_t)R * 512);
    float* bcb    = alc(R);
    float* ub     = alc((size_t)R * 512);

    auto gemm = [&](const float* A, int lda, const float* W, const float* bias,
                    const float* addp, int ldadd, float* Cp, int ldc,
                    int M, int N, int K) {
        dim3 g((N + 127) / 128, (M + 63) / 64);
#if HAS_ASYNC
        if ((K % 32 == 0) && (N % 128 == 0) && (lda % 4 == 0)) {
            gemm_wmma_kernel<true><<<g, 256, 0, stream>>>(A, lda, W, bias, addp, ldadd,
                                                          Cp, ldc, M, N, K);
            return;
        }
#endif
        gemm_wmma_kernel<false><<<g, 256, 0, stream>>>(A, lda, W, bias, addp, ldadd,
                                                       Cp, ldc, M, N, K);
    };
    auto ew = [](int n) { return dim3((n + 255) / 256); };

    auto mambaL1 = [&](const MP& m, const float* hin, int dm, float* dst, int dstld,
                       const float* addp, int ldadd) {
        int di  = dm;                 // expand = 1
        int dtr = (dm >= 16) ? dm / 16 : 1;
        int xwN = dtr + 32;
        gemm(hin, dm, m.in_w, nullptr, nullptr, 0, xzb, 2 * di, R, 2 * di, dm);
        mamba_pre_kernel<<<ew(R * di), 256, 0, stream>>>(xzb, m.conv_w, m.conv_b, xcb, R, di);
        gemm(xcb, di, m.x_w, nullptr, nullptr, 0, dblb, xwN, R, xwN, di);
        gemm(dblb, xwN, m.dt_w, m.dt_b, nullptr, 0, dtrawb, di, R, di, dtr);
        bcdot_kernel<<<ew(R), 256, 0, stream>>>(dblb, xwN, dtr, dtr + 16, bcb, R);
        mamba_post_kernel<<<ew(R * di), 256, 0, stream>>>(dtrawb, xcb, xzb, bcb, m.D, ub, R, di);
        gemm(ub, di, m.out_w, nullptr, addp, ldadd, dst, dstld, R, di, di);
    };
    auto mambaSeq = [&](const MP& m, const float* seq, int ldin, float* outp, int ldout,
                        int addmode, int Lseq) {
        mamba_seq_kernel<<<ew(R), 256, 0, stream>>>(seq, ldin, outp, ldout, addmode,
            m.in_w, m.conv_w, m.conv_b, m.x_w, m.dt_w, m.dt_b, m.A_log, m.D, m.out_w, R, Lseq);
    };

    normalize_kernel<<<R, 256, 0, stream>>>(X, xn, meanB, stdB, Bb, L, C);
    gemm(xn, 512, LW[1], LB[1], nullptr, 0, h1, 512, R, 512, 512);             // h1 = lin1
    mambaL1(M_[3], h1, 512, cat + 512, 2048, nullptr, 0);                      // x3 -> cat[512..]
    mambaSeq(M_[4], h1, 512, cat + 512, 2048, 1, 512);                         // += x4
    gemm(h1, 512, LW[2], LB[2], nullptr, 0, h2, 256, R, 256, 512);             // h2 = lin2
    mambaL1(M_[2], h2, 256, tmpA, 256, nullptr, 0);                            // x2
    mambaSeq(M_[1], h2, 256, tmpB, 256, 0, 256);                               // x1
    add3_kernel<<<ew(R * 256), 256, 0, stream>>>(tmpB, h2, tmpA, hsum, R * 256);
    gemm(hsum, 256, LW[3], LB[3], h1, 512, cat, 2048, R, 512, 256);            // lin3 + h1
    gemm(h2, 256, LW[5], LB[5], nullptr, 0, x5, 128, R, 128, 256);             // x5 = lin5
    mambaL1(M_[6], x5, 128, tmpA, 128, nullptr, 0);                            // x6
    mambaSeq(M_[5], x5, 128, tmpB, 128, 0, 128);                               // x5p
    add3_kernel<<<ew(R * 128), 256, 0, stream>>>(tmpB, x5, tmpA, s6, R * 128);
    gemm(s6, 128, LW[7], LB[7], nullptr, 0, cat + 1024, 2048, R, 512, 128);    // lin7
    gemm(x5, 128, LW[6], LB[6], nullptr, 0, x7, 16, R, 16, 128);               // x7 = lin6
    mambaL1(M_[8], x7, 16, tmpA, 16, nullptr, 0);                              // x8
    mambaSeq(M_[7], x7, 16, tmpB, 16, 0, 16);                                  // x7p
    add3_kernel<<<ew(R * 16), 256, 0, stream>>>(tmpB, x7, tmpA, s8, R * 16);
    gemm(s8, 16, LW[8], LB[8], nullptr, 0, cat + 1536, 2048, R, 512, 16);      // lin8
    gemm(cat, 2048, LW[4], LB[4], nullptr, 0, p4, 96, R, 96, 2048);            // lin4
    final_kernel<<<ew(Bb * P * C), 256, 0, stream>>>(p4, meanB, stdB, (float*)d_out, Bb, P, C);
}
